// CONVKNRM_87385404604464
// MI455X (gfx1250) — matrix operand order repaired and checked by reading, not compiled, not run on hardware
//
#include <hip/hip_runtime.h>
#include <hip/hip_bf16.h>
#include <math.h>

typedef __attribute__((ext_vector_type(16))) _Float16 v16h;
typedef __attribute__((ext_vector_type(8)))  _Float16 v8h;
typedef __attribute__((ext_vector_type(8)))  float    v8f;

#define B_    64
#define QLEN  20
#define DLEN  2048
#define E_    128
#define C_    128
#define NB_   11

static __device__ __forceinline__ v16h pack16(v8h lo, v8h hi) {
  v16h r;
#pragma unroll
  for (int i = 0; i < 8; ++i) { r[i] = lo[i]; r[i + 8] = hi[i]; }
  return r;
}

// ---------------------------------------------------------------------------
// Embedding gather + f32 -> f16 conversion
// ---------------------------------------------------------------------------
__global__ void gather_docs_f16(const int* __restrict__ idx,
                                const float* __restrict__ emb,
                                _Float16* __restrict__ out) {
  int t   = threadIdx.x;            // 0..127 (E dim)
  int pos = blockIdx.x % DLEN;
  int b   = blockIdx.x / DLEN;
  int tok = idx[b * DLEN + pos];
  out[((size_t)b * DLEN + pos) * E_ + t] =
      (_Float16)emb[(size_t)tok * E_ + t];
}

__global__ void gather_queries_f16(const int* __restrict__ idx,
                                   const float* __restrict__ emb,
                                   _Float16* __restrict__ out) {
  int t   = threadIdx.x;            // 0..127
  int pos = blockIdx.x % 32;        // padded to 32 rows
  int b   = blockIdx.x / 32;
  float v = 0.0f;
  if (pos < QLEN) {
    int tok = idx[b * QLEN + pos];
    v = emb[(size_t)tok * E_ + t];
  }
  out[((size_t)b * 32 + pos) * E_ + t] = (_Float16)v;
}

__global__ void cvt_f32_to_f16(const float* __restrict__ src,
                               _Float16* __restrict__ dst, int n) {
  int i = blockIdx.x * 256 + threadIdx.x;
  if (i < n) dst[i] = (_Float16)src[i];
}

__global__ void zero_f32(float* __restrict__ p, int n) {
  int i = blockIdx.x * 256 + threadIdx.x;
  if (i < n) p[i] = 0.0f;
}

// ---------------------------------------------------------------------------
// N-gram conv as WMMA GEMM with fused bias + ReLU + eps + L2-normalize.
// One wave per 16-row output tile. A = sliding embedding window (contiguous
// k*E halves per row), B = W stored [C][kE] f16. Output f16 [rows][C].
// Per K-step: one clause of 8x2 b128 loads, then 8 back-to-back WMMAs.
// ---------------------------------------------------------------------------
template <int KTOK>
__global__ __launch_bounds__(32) void conv_norm(
    const _Float16* __restrict__ eh,   // [B][Ltok][E] f16
    const _Float16* __restrict__ Wh,   // [C][KTOK*E] f16
    const float* __restrict__ bias,    // [C]
    _Float16* __restrict__ outp,       // [B][rowsTotal][C] f16 (normalized)
    int Ltok, int Lp, int Mtiles, int rowsTotal) {
  constexpr int KE     = KTOK * E_;
  constexpr int KSTEPS = KE / 32;

  int bb = blockIdx.x / Mtiles;
  int mt = blockIdx.x % Mtiles;
  int l  = threadIdx.x;
  int nn = l & 15;          // A-row (m) for loads / B-column (n) / D-column
  int h  = l >> 4;          // half-wave select

  int pos = mt * 16 + nn;
  int pc  = pos < (Lp - 1) ? pos : (Lp - 1);   // clamp: padded rows duplicate
  const _Float16* arow = eh + ((size_t)bb * Ltok + pc) * E_;
  const _Float16* wbase = Wh + (size_t)nn * KE + h * 8;

  v8f acc[8];
#pragma unroll
  for (int nt = 0; nt < 8; ++nt) acc[nt] = (v8f)0.0f;

#pragma unroll
  for (int ks = 0; ks < KSTEPS; ++ks) {
    int ck = ks * 32;
    v16h a = pack16(*(const v8h*)(arow + ck + h * 8),
                    *(const v8h*)(arow + ck + 16 + h * 8));
    v16h bfr[8];
#pragma unroll
    for (int nt = 0; nt < 8; ++nt) {     // batch all B loads into one clause
      const _Float16* wrow = wbase + (size_t)(nt * 16) * KE + ck;
      bfr[nt] = pack16(*(const v8h*)wrow, *(const v8h*)(wrow + 16));
    }
#pragma unroll
    for (int nt = 0; nt < 8; ++nt) {
      acc[nt] = __builtin_amdgcn_wmma_f32_16x16x32_f16(
          false, a, false, bfr[nt], (short)0, acc[nt], false, false);
    }
  }

  // Fused epilogue: bias + relu + 1e-9, then L2 norm over C=128.
  float ss[8];
#pragma unroll
  for (int p = 0; p < 8; ++p) ss[p] = 0.0f;
#pragma unroll
  for (int nt = 0; nt < 8; ++nt) {
    float bv = bias[nt * 16 + nn];
#pragma unroll
    for (int p = 0; p < 8; ++p) {
      float v = acc[nt][p] + bv;
      v = fmaxf(v, 0.0f) + 1e-9f;
      acc[nt][p] = v;
      ss[p] += v * v;
    }
  }
  // Row r = h*8+p is spread over the 16 lanes of each half: xor-reduce.
#pragma unroll
  for (int p = 0; p < 8; ++p) {
    ss[p] += __shfl_xor(ss[p], 1, 32);
    ss[p] += __shfl_xor(ss[p], 2, 32);
    ss[p] += __shfl_xor(ss[p], 4, 32);
    ss[p] += __shfl_xor(ss[p], 8, 32);
  }
  float rn[8];
#pragma unroll
  for (int p = 0; p < 8; ++p)
    rn[p] = 1.0f / fmaxf(sqrtf(ss[p]), 1e-10f);

#pragma unroll
  for (int nt = 0; nt < 8; ++nt) {
#pragma unroll
    for (int p = 0; p < 8; ++p) {
      int row = mt * 16 + h * 8 + p;
      outp[((size_t)bb * rowsTotal + row) * C_ + nt * 16 + nn] =
          (_Float16)(acc[nt][p] * rn[p]);
    }
  }
}

// ---------------------------------------------------------------------------
// Kernel pooling: sim = q x d^T via WMMA (K=C=128), fused Gaussian binning,
// sum over docs, log, sum over query rows, atomicAdd into feats[B][99].
// One wave per (batch, pair, q-tile); doc tiles are double-buffered so the
// next tile's 8 b128 loads overlap the 88-exp binning tail of this tile.
// ---------------------------------------------------------------------------
__global__ __launch_bounds__(32) void kernel_pool(
    const _Float16* __restrict__ qn0, const _Float16* __restrict__ qn1,
    const _Float16* __restrict__ qn2, const _Float16* __restrict__ dn0,
    const _Float16* __restrict__ dn1, const _Float16* __restrict__ dn2,
    const float* __restrict__ mu, const float* __restrict__ sigma,
    float* __restrict__ feats) {
  int idx  = blockIdx.x;
  int qt   = idx & 1;  idx >>= 1;
  int pair = idx % 9;
  int bb   = idx / 9;
  // pair order: uu, ut, ub, bu, tu, bb, bt, tb, tt  (2-bit packed LUTs)
  int qk = (0x29640 >> (2 * pair)) & 3;
  int dk = (0x26418 >> (2 * pair)) & 3;
  const _Float16* qn = (qk == 0) ? qn0 : ((qk == 1) ? qn1 : qn2);
  const _Float16* dn = (dk == 0) ? dn0 : ((dk == 1) ? dn1 : dn2);
  int Qp = QLEN - qk;
  int Dp = DLEN - dk;

  int l  = threadIdx.x;
  int nn = l & 15;
  int h  = l >> 4;

  // Per-bin Gaussian folded into exp2: contrib = exp2(abin * (s - mu)^2).
  // mu/abin stay uniform -> SGPRs.
  float mr[NB_], ab[NB_];
#pragma unroll
  for (int bin = 0; bin < NB_; ++bin) {
    mr[bin] = mu[bin];
    float rs = 1.0f / sigma[bin];
    ab[bin] = -0.72134752044f * rs * rs;   // -0.5 * log2(e) / sigma^2
  }

  // Lane-static row mask (replaces divergent exec-mask churn in inner loop).
  float rowm[8];
#pragma unroll
  for (int p = 0; p < 8; ++p)
    rowm[p] = ((qt * 16 + h * 8 + p) < Qp) ? 1.0f : 0.0f;

  // Query A fragments (K = C = 128 -> 4 steps), loaded once.
  const _Float16* qrow = qn + ((size_t)bb * 32 + qt * 16 + nn) * C_;
  v16h aq[4];
#pragma unroll
  for (int ks = 0; ks < 4; ++ks) {
    int ck = ks * 32;
    aq[ks] = pack16(*(const v8h*)(qrow + ck + h * 8),
                    *(const v8h*)(qrow + ck + 16 + h * 8));
  }

  float pl[8][NB_];
#pragma unroll
  for (int p = 0; p < 8; ++p)
#pragma unroll
    for (int bin = 0; bin < NB_; ++bin) pl[p][bin] = 0.0f;

  const _Float16* dbase = dn + ((size_t)bb * DLEN + nn) * C_ + h * 8;

  auto loadTile = [&](int dt, v16h bf[4]) {
    const _Float16* drow = dbase + (size_t)(dt * 16) * C_;
#pragma unroll
    for (int ks = 0; ks < 4; ++ks) {
      const _Float16* dp = drow + ks * 32;
      bf[ks] = pack16(*(const v8h*)dp, *(const v8h*)(dp + 16));
    }
  };

  auto compTile = [&](int dt, const v16h bf[4]) {
    v8f s = (v8f)0.0f;
#pragma unroll
    for (int ks = 0; ks < 4; ++ks)
      s = __builtin_amdgcn_wmma_f32_16x16x32_f16(
          false, aq[ks], false, bf[ks], (short)0, s, false, false);
    float cm = ((dt * 16 + nn) < Dp) ? 1.0f : 0.0f;  // doc-column mask
#pragma unroll
    for (int p = 0; p < 8; ++p) {
      float m  = cm * rowm[p];
      float sv = s[p];
#pragma unroll
      for (int bin = 0; bin < NB_; ++bin) {
        float t = sv - mr[bin];
        pl[p][bin] += m * __builtin_amdgcn_exp2f(t * t * ab[bin]);
      }
    }
  };

  v16h buf0[4], buf1[4];
  loadTile(0, buf0);
  for (int dt = 0; dt < DLEN / 16; dt += 2) {
    loadTile(dt + 1, buf1);          // in flight during buf0's bin tail
    compTile(dt, buf0);
    if (dt + 2 < DLEN / 16) loadTile(dt + 2, buf0);
    compTile(dt + 1, buf1);
  }

  // Sum over doc columns held by the 16 lanes of each half.
#pragma unroll
  for (int p = 0; p < 8; ++p) {
#pragma unroll
    for (int bin = 0; bin < NB_; ++bin) {
      float v = pl[p][bin];
      v += __shfl_xor(v, 1, 32);
      v += __shfl_xor(v, 2, 32);
      v += __shfl_xor(v, 4, 32);
      v += __shfl_xor(v, 8, 32);
      pl[p][bin] = v;
    }
  }

  // log(clip) * 0.01, masked-summed over this tile's valid query rows.
  float ls[NB_];
#pragma unroll
  for (int bin = 0; bin < NB_; ++bin) ls[bin] = 0.0f;
#pragma unroll
  for (int p = 0; p < 8; ++p) {
#pragma unroll
    for (int bin = 0; bin < NB_; ++bin)
      ls[bin] += rowm[p] * __logf(fmaxf(pl[p][bin], 1e-10f)) * 0.01f;
  }
#pragma unroll
  for (int bin = 0; bin < NB_; ++bin)
    ls[bin] += __shfl_xor(ls[bin], 16, 32);   // combine the two halves

  if (l == 0) {
#pragma unroll
    for (int bin = 0; bin < NB_; ++bin)
      atomicAdd(&feats[(size_t)bb * (9 * NB_) + pair * NB_ + bin], ls[bin]);
  }
}

// ---------------------------------------------------------------------------
// Final dense [99 -> 1] + tanh.
// ---------------------------------------------------------------------------
__global__ void dense_tanh(const float* __restrict__ feats,
                           const float* __restrict__ w,
                           const float* __restrict__ b0,
                           float* __restrict__ out) {
  int bb = threadIdx.x;  // 64 threads
  float s = b0[0];
#pragma unroll 9
  for (int i = 0; i < 9 * NB_; ++i) s += feats[(size_t)bb * (9 * NB_) + i] * w[i];
  out[bb] = tanhf(s);
}

// ---------------------------------------------------------------------------
extern "C" void kernel_launch(void* const* d_in, const int* in_sizes, int n_in,
                              void* d_out, int out_size, void* d_ws,
                              size_t ws_size, hipStream_t stream) {
  (void)in_sizes; (void)n_in; (void)out_size; (void)ws_size;
  const int*   q_idx   = (const int*)d_in[0];
  const int*   d_idx   = (const int*)d_in[1];
  // d_in[2] = batch_semantic (unused by the reference output path)
  const float* emb     = (const float*)d_in[3];
  const float* W_uni   = (const float*)d_in[4];
  const float* b_uni   = (const float*)d_in[5];
  const float* W_bi    = (const float*)d_in[6];
  const float* b_bi    = (const float*)d_in[7];
  const float* W_tri   = (const float*)d_in[8];
  const float* b_tri   = (const float*)d_in[9];
  const float* dense_w = (const float*)d_in[10];
  const float* dense_b = (const float*)d_in[11];
  const float* mu      = (const float*)d_in[12];
  const float* sigma   = (const float*)d_in[13];
  float* out = (float*)d_out;

  // Workspace carve-up (256B aligned chunks).
  char* ws = (char*)d_ws;
  size_t off = 0;
  auto take = [&](size_t bytes) {
    char* p = ws + off;
    off += (bytes + 255) & ~(size_t)255;
    return p;
  };
  _Float16* edh = (_Float16*)take((size_t)B_ * DLEN * E_ * 2);
  _Float16* eqh = (_Float16*)take((size_t)B_ * 32 * E_ * 2);
  _Float16* Wh0 = (_Float16*)take((size_t)C_ * 1 * E_ * 2);
  _Float16* Wh1 = (_Float16*)take((size_t)C_ * 2 * E_ * 2);
  _Float16* Wh2 = (_Float16*)take((size_t)C_ * 3 * E_ * 2);
  _Float16* qn0 = (_Float16*)take((size_t)B_ * 32 * C_ * 2);
  _Float16* qn1 = (_Float16*)take((size_t)B_ * 32 * C_ * 2);
  _Float16* qn2 = (_Float16*)take((size_t)B_ * 32 * C_ * 2);
  _Float16* dn0 = (_Float16*)take((size_t)B_ * DLEN * C_ * 2);
  _Float16* dn1 = (_Float16*)take((size_t)B_ * DLEN * C_ * 2);
  _Float16* dn2 = (_Float16*)take((size_t)B_ * DLEN * C_ * 2);
  float*    feats = (float*)take((size_t)B_ * 9 * NB_ * 4);

  // 1) init + weight conversion + gathers
  zero_f32<<<(B_ * 9 * NB_ + 255) / 256, 256, 0, stream>>>(feats, B_ * 9 * NB_);
  cvt_f32_to_f16<<<(C_ * 1 * E_ + 255) / 256, 256, 0, stream>>>(W_uni, Wh0, C_ * 1 * E_);
  cvt_f32_to_f16<<<(C_ * 2 * E_ + 255) / 256, 256, 0, stream>>>(W_bi,  Wh1, C_ * 2 * E_);
  cvt_f32_to_f16<<<(C_ * 3 * E_ + 255) / 256, 256, 0, stream>>>(W_tri, Wh2, C_ * 3 * E_);
  gather_queries_f16<<<B_ * 32, 128, 0, stream>>>(q_idx, emb, eqh);
  gather_docs_f16<<<B_ * DLEN, 128, 0, stream>>>(d_idx, emb, edh);

  // 2) conv + relu + l2norm (WMMA GEMM, fused epilogue)
  conv_norm<1><<<B_ * 2,   32, 0, stream>>>(eqh, Wh0, b_uni, qn0, 32,   QLEN,     2,   32);
  conv_norm<2><<<B_ * 2,   32, 0, stream>>>(eqh, Wh1, b_bi,  qn1, 32,   QLEN - 1, 2,   32);
  conv_norm<3><<<B_ * 2,   32, 0, stream>>>(eqh, Wh2, b_tri, qn2, 32,   QLEN - 2, 2,   32);
  conv_norm<1><<<B_ * 128, 32, 0, stream>>>(edh, Wh0, b_uni, dn0, DLEN, DLEN,     128, DLEN);
  conv_norm<2><<<B_ * 128, 32, 0, stream>>>(edh, Wh1, b_bi,  dn1, DLEN, DLEN - 1, 128, DLEN);
  conv_norm<3><<<B_ * 128, 32, 0, stream>>>(edh, Wh2, b_tri, dn2, DLEN, DLEN - 2, 128, DLEN);

  // 3) similarity (WMMA) + Gaussian kernel pooling + log-sum
  kernel_pool<<<B_ * 9 * 2, 32, 0, stream>>>(qn0, qn1, qn2, dn0, dn1, dn2,
                                             mu, sigma, feats);

  // 4) dense + tanh
  dense_tanh<<<1, 64, 0, stream>>>(feats, dense_w, dense_b, out);
}
